// FastKANLayer_47579647705728
// MI455X (gfx1250) — compile-verified, matrix operand order checked by reading, not run on hardware
//
#include <hip/hip_runtime.h>
#include <hip/hip_bf16.h>

typedef __bf16 bf16;
typedef __bf16 bf16x8 __attribute__((ext_vector_type(8)));
typedef __bf16 v16bf  __attribute__((ext_vector_type(16)));
typedef float  v8f    __attribute__((ext_vector_type(8)));
typedef int    v4i_   __attribute__((vector_size(16)));

#define IN_DIM  1024
#define OUT_DIM 1024
#define NC      35
#define KSPL    (IN_DIM * NC)        // 35840
#define KTOT    (KSPL + IN_DIM)      // 36864, multiple of 32
#define BK      32
#define NKTILE  (KTOT / BK)          // 1152, multiple of 3
#define LDS_K   40                   // 32 + 8 bf16 pad -> 20-dword stride, conflict-free

static_assert(NKTILE % 3 == 0 && NKTILE >= 6, "pipeline needs nK % 3 == 0");

// ---- CDNA5 async global->LDS staging (ASYNCcnt path) ----------------------
#if defined(__gfx1250__) && __has_builtin(__builtin_amdgcn_global_load_async_to_lds_b128)
#define USE_ASYNC_LDS 1
#else
#define USE_ASYNC_LDS 0
#endif

__device__ __forceinline__ void wait_async0() {
#if __has_builtin(__builtin_amdgcn_s_wait_asynccnt)
    __builtin_amdgcn_s_wait_asynccnt(0);
#else
    asm volatile("s_wait_asynccnt 0x0" ::: "memory");
#endif
}
__device__ __forceinline__ void wait_async4() {
#if __has_builtin(__builtin_amdgcn_s_wait_asynccnt)
    __builtin_amdgcn_s_wait_asynccnt(4);
#else
    asm volatile("s_wait_asynccnt 0x4" ::: "memory");
#endif
}

#if USE_ASYNC_LDS
__device__ __forceinline__ void async_cp16(const void* g, void* l) {
    // signature (from clang diag): (AS1 v4i*, AS3 v4i*, int offset, int cpol)
    __builtin_amdgcn_global_load_async_to_lds_b128(
        (__attribute__((address_space(1))) v4i_*)(g),
        (__attribute__((address_space(3))) v4i_*)(l),
        /*offset=*/0, /*cpol=*/0);
}
#endif

// ---------------------------------------------------------------------------
// Kernel 1: build A[b, k] in bf16.
//   k <  KSPL : cubic B-spline basis of clip(x[b, k/35]) at grid point k%35
//   k >= KSPL : silu(clip(x[b, k-KSPL]))   (folds the base GEMM into K)
// ---------------------------------------------------------------------------
__global__ __launch_bounds__(256)
void prep_basis(const float* __restrict__ x, bf16* __restrict__ A) {
    __shared__ float xs[IN_DIM];
    const int b = blockIdx.x;
    for (int i = threadIdx.x; i < IN_DIM; i += 256) {
        float v = x[(size_t)b * IN_DIM + i];
        xs[i] = fminf(fmaxf(v, -0.99f), 0.99f);
    }
    __syncthreads();

    bf16* out = A + (size_t)b * KTOT;
    for (int k = threadIdx.x; k < KTOT; k += 256) {
        float v;
        if (k < KSPL) {
            const int i = k / NC;
            const int c = k - i * NC;
            const float gc = -1.0f + (float)c * (2.0f / 34.0f);
            const float d  = fabsf(xs[i] - gc) * 17.0f;   // /h, h = 2/34
            if (d < 1.0f) {
                v = 0.66666667f - d * d + 0.5f * d * d * d;
            } else if (d < 2.0f) {
                const float t = 2.0f - d;
                v = t * t * t * (1.0f / 6.0f);
            } else {
                v = 0.0f;
            }
        } else {
            const float xv = xs[k - KSPL];
            v = xv / (1.0f + __expf(-xv));                // silu
        }
        out[k] = (bf16)v;
    }
}

// ---------------------------------------------------------------------------
// Kernel 2: build Bt[j, k] in bf16 (GEMM B transposed: k contiguous per row).
//   k <  KSPL : spline_weight[i, j, c] with k = i*35 + c
//   k >= KSPL : base_scale[k-KSPL, j]
// One block per j -> writes fully coalesced.
// ---------------------------------------------------------------------------
__global__ __launch_bounds__(256)
void prep_weight(const float* __restrict__ sw, const float* __restrict__ bsc,
                 bf16* __restrict__ Bt) {
    const int j = blockIdx.x;
    bf16* out = Bt + (size_t)j * KTOT;
    for (int k = threadIdx.x; k < KTOT; k += 256) {
        float w;
        if (k < KSPL) {
            const int i = k / NC;
            const int c = k - i * NC;
            w = sw[((size_t)i * OUT_DIM + j) * NC + c];
        } else {
            w = bsc[(size_t)(k - KSPL) * OUT_DIM + j];
        }
        out[k] = (bf16)w;
    }
}

// ---------------------------------------------------------------------------
// Kernel 3: C[M, 1024] f32 = A[M, KTOT] * Bt[1024, KTOT]^T  (bf16 WMMA, f32 acc)
// Block tile 128x128, 8 waves = 4 (M) x 2 (N); wave tile 32x64 = 2x4 frags.
// BK = 32 per step. Async global->LDS DMA (ASYNCcnt), 3 LDS slots, TWO tiles
// in flight. Steady-state loop processes 3 tiles with CONSTANT slots and
// constant waits (s_wait_asynccnt 4) -> branch-free hot loop; last 3 tiles
// peeled. Fallback = reg-prefetch double buffer.
// ---------------------------------------------------------------------------
__global__ __launch_bounds__(256)
void kan_gemm(const bf16* __restrict__ A, const bf16* __restrict__ Bt,
              float* __restrict__ C) {
    __shared__ bf16 As[3][128][LDS_K];
    __shared__ bf16 Bs[3][128][LDS_K];

    const int tid  = threadIdx.x;
    const int row0 = blockIdx.y * 128;   // M tile
    const int col0 = blockIdx.x * 128;   // N tile

    const int wave = tid >> 5;
    const int lane = tid & 31;
    const int wm   = wave >> 1;          // 0..3 -> 32 rows each
    const int wn   = wave & 1;           // 0..1 -> 64 cols each
    const int lr   = lane & 15;
    const int lh   = lane >> 4;

    // tile loaders: each thread moves 2 x 16B per operand per step
    const int ldr = tid >> 2;            // 0..63
    const int ldk = (tid & 3) * 8;       // 0,8,16,24

    const bf16* gA0 = A  + (size_t)(row0 + ldr)      * KTOT + ldk;
    const bf16* gA1 = A  + (size_t)(row0 + ldr + 64) * KTOT + ldk;
    const bf16* gB0 = Bt + (size_t)(col0 + ldr)      * KTOT + ldk;
    const bf16* gB1 = Bt + (size_t)(col0 + ldr + 64) * KTOT + ldk;

    v8f acc[2][4];
    const v8f vzero = {0.f, 0.f, 0.f, 0.f, 0.f, 0.f, 0.f, 0.f};
#pragma unroll
    for (int mi = 0; mi < 2; ++mi)
#pragma unroll
        for (int ni = 0; ni < 4; ++ni)
            acc[mi][ni] = vzero;

    // Compute one BK-slab from LDS slot (slot constant at every call site).
    auto compute_slot = [&](int slot) {
        v16bf af[2], bfr[4];
#pragma unroll
        for (int mi = 0; mi < 2; ++mi) {
            const int r = wm * 32 + mi * 16 + lr;
            bf16x8 lo = *(const bf16x8*)&As[slot][r][lh * 8];
            bf16x8 hi = *(const bf16x8*)&As[slot][r][lh * 8 + 16];
            af[mi] = __builtin_shufflevector(lo, hi, 0, 1, 2, 3, 4, 5, 6, 7,
                                             8, 9, 10, 11, 12, 13, 14, 15);
        }
#pragma unroll
        for (int ni = 0; ni < 4; ++ni) {
            const int r = wn * 64 + ni * 16 + lr;
            bf16x8 lo = *(const bf16x8*)&Bs[slot][r][lh * 8];
            bf16x8 hi = *(const bf16x8*)&Bs[slot][r][lh * 8 + 16];
            bfr[ni] = __builtin_shufflevector(lo, hi, 0, 1, 2, 3, 4, 5, 6, 7,
                                              8, 9, 10, 11, 12, 13, 14, 15);
        }
#pragma unroll
        for (int mi = 0; mi < 2; ++mi)
#pragma unroll
            for (int ni = 0; ni < 4; ++ni)
                acc[mi][ni] = __builtin_amdgcn_wmma_f32_16x16x32_bf16(
                    false, af[mi], false, bfr[ni], (short)0, acc[mi][ni],
                    false, false);
    };

#if USE_ASYNC_LDS
    auto dma_tile = [&](int t, int slot) {   // 4 async b128 DMAs per thread
        const size_t off = (size_t)t * BK;
        async_cp16(gA0 + off, &As[slot][ldr][ldk]);
        async_cp16(gA1 + off, &As[slot][ldr + 64][ldk]);
        async_cp16(gB0 + off, &Bs[slot][ldr][ldk]);
        async_cp16(gB1 + off, &Bs[slot][ldr + 64][ldk]);
    };

    // prologue: tiles 0,1 in flight; retire tile 0 (in-order completion)
    dma_tile(0, 0);
    dma_tile(1, 1);
    wait_async4();
    __syncthreads();

    // steady state: 3 tiles per iteration, constant slots, constant waits.
    // body at kt: DMA kt+2..kt+4 while computing kt..kt+2.
    for (int kt = 0; kt + 6 <= NKTILE; kt += 3) {
        dma_tile(kt + 2, 2); compute_slot(0); wait_async4(); __syncthreads();
        dma_tile(kt + 3, 0); compute_slot(1); wait_async4(); __syncthreads();
        dma_tile(kt + 4, 1); compute_slot(2); wait_async4(); __syncthreads();
    }
    // tail: tiles nK-3 (slot0), nK-2 (slot1), nK-1 (slot2); nK-1 not yet DMA'd
    dma_tile(NKTILE - 1, 2);
    compute_slot(0); wait_async4(); __syncthreads();
    compute_slot(1); wait_async0(); __syncthreads();
    compute_slot(2);
#else
    uint4 ra0 = *(const uint4*)gA0;
    uint4 ra1 = *(const uint4*)gA1;
    uint4 rb0 = *(const uint4*)gB0;
    uint4 rb1 = *(const uint4*)gB1;
    int cur = 0;
    for (int kt = 0; kt < NKTILE; ++kt) {
        *(uint4*)&As[cur][ldr     ][ldk] = ra0;
        *(uint4*)&As[cur][ldr + 64][ldk] = ra1;
        *(uint4*)&Bs[cur][ldr     ][ldk] = rb0;
        *(uint4*)&Bs[cur][ldr + 64][ldk] = rb1;
        __syncthreads();
        if (kt + 1 < NKTILE) {
            const size_t off = (size_t)(kt + 1) * BK;
            ra0 = *(const uint4*)(gA0 + off);
            ra1 = *(const uint4*)(gA1 + off);
            rb0 = *(const uint4*)(gB0 + off);
            rb1 = *(const uint4*)(gB1 + off);
        }
        compute_slot(cur);
        __syncthreads();
        cur ^= 1;
    }
#endif

    // Epilogue: C/D layout -> lane lr = col, lanes 16..31 cover rows +8,
    // vector element r = row offset r.
#pragma unroll
    for (int mi = 0; mi < 2; ++mi) {
#pragma unroll
        for (int ni = 0; ni < 4; ++ni) {
            const int col = col0 + wn * 64 + ni * 16 + lr;
            const int rowb = row0 + wm * 32 + mi * 16 + lh * 8;
#pragma unroll
            for (int r = 0; r < 8; ++r)
                C[(size_t)(rowb + r) * OUT_DIM + col] = acc[mi][ni][r];
        }
    }
}

// ---------------------------------------------------------------------------
extern "C" void kernel_launch(void* const* d_in, const int* in_sizes, int n_in,
                              void* d_out, int out_size, void* d_ws, size_t ws_size,
                              hipStream_t stream) {
    const float* x   = (const float*)d_in[0];   // [M, 1024]
    const float* sw  = (const float*)d_in[1];   // [1024, 1024, 35]
    const float* bsc = (const float*)d_in[2];   // [1024, 1024]
    // d_in[3] (grid) is analytic linspace(-1,1,35); recomputed on device.

    const int M = in_sizes[0] / IN_DIM;         // 4096

    bf16* Abf = (bf16*)d_ws;
    const size_t Abytes = (size_t)M * KTOT * sizeof(bf16);   // 302 MB, 16B-aligned
    bf16* Btbf = (bf16*)((char*)d_ws + Abytes);              // 75 MB

    prep_basis<<<M, 256, 0, stream>>>(x, Abf);
    prep_weight<<<OUT_DIM, 256, 0, stream>>>(sw, bsc, Btbf);
    kan_gemm<<<dim3(OUT_DIM / 128, M / 128), 256, 0, stream>>>(Abf, Btbf,
                                                               (float*)d_out);
}
